// MultiHeadSelfAttention_80470507258333
// MI455X (gfx1250) — matrix operand order repaired and checked by reading, not compiled, not run on hardware
//
#include <hip/hip_runtime.h>

// ---------------------------------------------------------------------------
// Problem constants (fixed by the reference)
// ---------------------------------------------------------------------------
constexpr int BS  = 4;
constexpr int SEQ = 2048;
constexpr int DM  = 1024;
constexpr int NH  = 16;
constexpr int HD  = 64;   // head dim
constexpr int R   = BS * SEQ;  // 8192 rows

typedef _Float16 v8h  __attribute__((ext_vector_type(8)));
typedef _Float16 v16h __attribute__((ext_vector_type(16)));
typedef float    v8f  __attribute__((ext_vector_type(8)));
typedef int      v4i  __attribute__((ext_vector_type(4)));

// Address-space qualified int4 pointers for the async global->LDS builtin:
// param0 = global (AS1) int4*, param1 = LDS (AS3) int4*.
typedef __attribute__((address_space(1))) v4i gas_v4i;
typedef __attribute__((address_space(3))) v4i las_v4i;

#if __has_builtin(__builtin_amdgcn_global_load_async_to_lds_b128)
#define HAVE_ASYNC_LDS 1
#else
#define HAVE_ASYNC_LDS 0
#endif

__device__ __forceinline__ void wait_asynccnt0() {
#if __has_builtin(__builtin_amdgcn_s_wait_asynccnt)
  __builtin_amdgcn_s_wait_asynccnt(0);
#else
  asm volatile("s_wait_asynccnt 0x0" ::: "memory");
#endif
}

__device__ __forceinline__ v16h cat8(v8h lo, v8h hi) {
  return __builtin_shufflevector(lo, hi, 0,1,2,3,4,5,6,7,8,9,10,11,12,13,14,15);
}

__device__ __forceinline__ v8f wmma_f16(v16h a, v16h b, v8f c) {
  // D = A(16x32 f16) * B(32x16 f16) + C(16x16 f32)   [wave32]
  return __builtin_amdgcn_wmma_f32_16x16x32_f16(
      /*neg_a=*/false, a, /*neg_b=*/false, b,
      /*c_mod=*/(short)0, c, /*reuse_a=*/false, /*reuse_b=*/false);
}

// ---------------------------------------------------------------------------
// f32 -> f16 conversion
// ---------------------------------------------------------------------------
__global__ void f32_to_f16_kernel(const float* __restrict__ in,
                                  _Float16* __restrict__ out, int n) {
  int i = blockIdx.x * blockDim.x + threadIdx.x;
  if (i < n) out[i] = (_Float16)in[i];
}

// ---------------------------------------------------------------------------
// GEMM: out[M,N] = A[M,K] * W[N,K]^T   (A,W f16; out f16 or f32)
// Block = 128 threads (4 waves), block tile 128x64, wave tile 32x64.
// Fragments loaded directly from global (working set is L2-resident on the
// 192MB L2 of MI455X); global_prefetch covers latency 4 K-steps ahead.
// ---------------------------------------------------------------------------
template <bool OUT_F32>
__global__ __launch_bounds__(128)
void gemm_wmma_kernel(const _Float16* __restrict__ A,
                      const _Float16* __restrict__ W,
                      void* __restrict__ out, int M, int N, int K) {
  const int lane = threadIdx.x & 31;
  const int wid  = threadIdx.x >> 5;
  const int ln   = lane & 15;
  const int hi8  = lane >> 4;
  const int row0 = blockIdx.x * 128 + wid * 32;
  const int col0 = blockIdx.y * 64;

  v8f acc[2][4] = {};

  for (int k0 = 0; k0 < K; k0 += 32) {
    if (k0 + 128 < K) {  // uniform condition: EXEC stays full
      __builtin_prefetch(A + (size_t)(row0 + ln) * K + k0 + 128, 0, 1);
      __builtin_prefetch(W + (size_t)(col0 + ln) * K + k0 + 128, 0, 1);
    }
    v16h a[2];
#pragma unroll
    for (int mi = 0; mi < 2; ++mi) {
      const _Float16* ap = A + (size_t)(row0 + mi * 16 + ln) * K + k0 + hi8 * 8;
      a[mi] = cat8(*(const v8h*)ap, *(const v8h*)(ap + 16));
    }
#pragma unroll
    for (int ni = 0; ni < 4; ++ni) {
      const _Float16* bp = W + (size_t)(col0 + ni * 16 + ln) * K + k0 + hi8 * 16;
      v16h b = cat8(*(const v8h*)(bp - hi8 * 8), *(const v8h*)(bp - hi8 * 8 + 16));
#pragma unroll
      for (int mi = 0; mi < 2; ++mi)
        acc[mi][ni] = wmma_f16(a[mi], b, acc[mi][ni]);
    }
  }

#pragma unroll
  for (int mi = 0; mi < 2; ++mi)
#pragma unroll
    for (int ni = 0; ni < 4; ++ni)
#pragma unroll
      for (int r = 0; r < 8; ++r) {
        const int row = row0 + mi * 16 + r + 8 * hi8;
        const int col = col0 + ni * 16 + ln;
        if constexpr (OUT_F32)
          ((float*)out)[(size_t)row * N + col] = acc[mi][ni][r];
        else
          ((_Float16*)out)[(size_t)row * N + col] = (_Float16)acc[mi][ni][r];
      }
}

// ---------------------------------------------------------------------------
// RoPE (in place) on Q or K, layout (BS*SEQ, NH*HD). One thread per pair.
// ---------------------------------------------------------------------------
__global__ void rope_kernel(_Float16* __restrict__ X,
                            const int* __restrict__ pos, int n) {
  const int idx = blockIdx.x * blockDim.x + threadIdx.x;
  if (idx >= n) return;
  const int i2 = idx & 31;           // pair index, HD/2 = 32
  const int h  = (idx >> 5) & (NH - 1);
  const int s  = (idx >> 9) & (SEQ - 1);
  const int b  = idx >> 20;
  const float p        = (float)pos[s];
  const float inv_freq = __powf(10000.f, -(2.f * (float)i2) / (float)HD);
  const float ang = p * inv_freq;
  const float cs = __cosf(ang), sn = __sinf(ang);
  const size_t base = (size_t)(b * SEQ + s) * DM + h * HD + 2 * i2;
  const float xe = (float)X[base], xo = (float)X[base + 1];
  X[base]     = (_Float16)(xe * cs - xo * sn);
  X[base + 1] = (_Float16)(xo * cs + xe * sn);
}

// ---------------------------------------------------------------------------
// Causal flash attention. Grid (SEQ/64, NH, BS), block 128 (4 waves).
// Each block: 64 query rows of one (b,h); each wave owns 16 query rows.
// K tile staged via async global->LDS copies (ASYNCcnt path) when available;
// V tile transposed through VGPRs; online softmax in f32.
// ---------------------------------------------------------------------------
__global__ __launch_bounds__(128)
void flash_attn_kernel(const _Float16* __restrict__ Q,
                       const _Float16* __restrict__ Kc,
                       const _Float16* __restrict__ V,
                       _Float16* __restrict__ O) {
  __shared__ _Float16 Ks[64 * 64];   // [key][dim]
  __shared__ _Float16 Vt[64 * 64];   // [dim][key] (transposed)
  __shared__ _Float16 Ps[64 * 64];   // wave-private P staging (16 rows/wave)

  const int qtile = blockIdx.x, h = blockIdx.y, b = blockIdx.z;
  const int tid = threadIdx.x, wid = tid >> 5, lane = tid & 31;
  const int ln = lane & 15, hi8 = lane >> 4;
  const int qbase = qtile * 64;

  // Q fragments held in registers for the whole kernel (K dim = 64 -> 2 steps)
  v16h qa[2];
  {
    const _Float16* qp =
        Q + (size_t)(b * SEQ + qbase + wid * 16 + ln) * DM + h * HD;
#pragma unroll
    for (int ks = 0; ks < 2; ++ks) {
      const _Float16* p = qp + ks * 32 + hi8 * 8;
      qa[ks] = cat8(*(const v8h*)p, *(const v8h*)(p + 16));
    }
  }

  v8f o[4] = {};
  float m_run[8], l_run[8];
#pragma unroll
  for (int r = 0; r < 8; ++r) { m_run[r] = -__builtin_inff(); l_run[r] = 0.f; }

  const int srow = tid >> 1, shalf = tid & 1;  // staging: row, 32-elem half

  for (int kt = 0; kt <= qtile; ++kt) {
    __syncthreads();  // previous tile's LDS reads complete
    {
      const size_t gro =
          (size_t)(b * SEQ + kt * 64 + srow) * DM + h * HD + shalf * 32;
      const _Float16* krp = Kc + gro;
#if HAVE_ASYNC_LDS
#pragma unroll
      for (int i = 0; i < 4; ++i)
        __builtin_amdgcn_global_load_async_to_lds_b128(
            (gas_v4i*)(krp + i * 8),
            (las_v4i*)(&Ks[srow * 64 + shalf * 32 + i * 8]),
            /*offset=*/0, /*cpol=*/0);
#else
#pragma unroll
      for (int i = 0; i < 4; ++i)
        *(v8h*)(&Ks[srow * 64 + shalf * 32 + i * 8]) = *(const v8h*)(krp + i * 8);
#endif
      const _Float16* vrp = V + gro;
#pragma unroll
      for (int j = 0; j < 32; ++j)
        Vt[(shalf * 32 + j) * 64 + srow] = vrp[j];
    }
#if HAVE_ASYNC_LDS
    wait_asynccnt0();
#endif
    __syncthreads();

    // S = Q * K^T  (wave's 16 rows x 64 keys)
    v8f s[4] = {};
#pragma unroll
    for (int ks = 0; ks < 2; ++ks)
#pragma unroll
      for (int ni = 0; ni < 4; ++ni) {
        v16h bv = cat8(*(const v8h*)(&Ks[(ni * 16 + ln) * 64 + ks * 32 + hi8 * 8]), *(const v8h*)(&Ks[(ni * 16 + ln) * 64 + ks * 32 + hi8 * 8 + 16]));
        s[ni] = wmma_f16(qa[ks], bv, s[ni]);
      }

    const bool diag = (kt == qtile);
#pragma unroll
    for (int r = 0; r < 8; ++r) {
      const int qg = qbase + wid * 16 + r + 8 * hi8;
      float mrow = -__builtin_inff();
#pragma unroll
      for (int ni = 0; ni < 4; ++ni) {
        float sv = s[ni][r] * 0.125f;  // 1/sqrt(64)
        if (diag && (kt * 64 + ni * 16 + ln) > qg) sv = -__builtin_inff();
        s[ni][r] = sv;
        mrow = fmaxf(mrow, sv);
      }
#pragma unroll
      for (int mk = 1; mk <= 8; mk <<= 1)      // reduce across 16-lane group
        mrow = fmaxf(mrow, __shfl_xor(mrow, mk, 32));
      const float mnew = fmaxf(m_run[r], mrow);
      float rs = 0.f;
#pragma unroll
      for (int ni = 0; ni < 4; ++ni) {
        const float p = __expf(s[ni][r] - mnew);
        s[ni][r] = p;
        rs += p;
      }
#pragma unroll
      for (int mk = 1; mk <= 8; mk <<= 1)
        rs += __shfl_xor(rs, mk, 32);
      const float alpha = __expf(m_run[r] - mnew);
      l_run[r] = l_run[r] * alpha + rs;
      m_run[r] = mnew;
#pragma unroll
      for (int ni = 0; ni < 4; ++ni) o[ni][r] *= alpha;
    }

    // P: C-layout f32 regs -> wave-private LDS -> A-layout f16 frags.
    // Same-wave LDS ops are in-order (DScnt), no workgroup barrier needed.
#pragma unroll
    for (int ni = 0; ni < 4; ++ni)
#pragma unroll
      for (int r = 0; r < 8; ++r)
        Ps[(wid * 16 + r + 8 * hi8) * 64 + ni * 16 + ln] = (_Float16)s[ni][r];

#pragma unroll
    for (int ks = 0; ks < 2; ++ks) {
      const _Float16* pp = &Ps[(wid * 16 + ln) * 64 + ks * 32 + hi8 * 8];
      v16h pa = cat8(*(const v8h*)pp, *(const v8h*)(pp + 16));
#pragma unroll
      for (int ni = 0; ni < 4; ++ni) {
        v16h bv = cat8(*(const v8h*)(&Vt[(ni * 16 + ln) * 64 + ks * 32 + hi8 * 8]), *(const v8h*)(&Vt[(ni * 16 + ln) * 64 + ks * 32 + hi8 * 8 + 16]));
        o[ni] = wmma_f16(pa, bv, o[ni]);
      }
    }
  }

  // epilogue: normalize, store f16 heads-interleaved (so out-proj is a GEMM)
  _Float16* op = O + (size_t)(b * SEQ + qbase + wid * 16) * DM + h * HD;
#pragma unroll
  for (int r = 0; r < 8; ++r) {
    const float il = 1.f / l_run[r];
#pragma unroll
    for (int ni = 0; ni < 4; ++ni)
      op[(size_t)(r + 8 * hi8) * DM + ni * 16 + ln] = (_Float16)(o[ni][r] * il);
  }
}

// ---------------------------------------------------------------------------
// Host launch
// ---------------------------------------------------------------------------
extern "C" void kernel_launch(void* const* d_in, const int* in_sizes, int n_in,
                              void* d_out, int out_size, void* d_ws, size_t ws_size,
                              hipStream_t stream) {
  const float* x  = (const float*)d_in[0];
  const float* Wq = (const float*)d_in[1];
  const float* Wk = (const float*)d_in[2];
  const float* Wv = (const float*)d_in[3];
  const float* Wo = (const float*)d_in[4];
  const int*   tp = (const int*)d_in[5];

  char* ws = (char*)d_ws;
  const size_t XH  = (size_t)R * DM * sizeof(_Float16);   // 16 MB
  const size_t WH  = (size_t)DM * DM * sizeof(_Float16);  //  2 MB
  _Float16* xh  = (_Float16*)(ws);
  _Float16* Wqh = (_Float16*)(ws + XH);
  _Float16* Wkh = (_Float16*)(ws + XH + WH);
  _Float16* Wvh = (_Float16*)(ws + XH + 2 * WH);
  _Float16* Woh = (_Float16*)(ws + XH + 3 * WH);
  _Float16* Qh  = (_Float16*)(ws + XH + 4 * WH);
  _Float16* Kh  = (_Float16*)(ws + 2 * XH + 4 * WH);
  _Float16* Vh  = (_Float16*)(ws + 3 * XH + 4 * WH);
  _Float16* Ah  = (_Float16*)(ws + 4 * XH + 4 * WH);

  // 1) convert inputs to f16
  {
    const int nx = R * DM;
    f32_to_f16_kernel<<<(nx + 255) / 256, 256, 0, stream>>>(x, xh, nx);
    const int nw = DM * DM;
    f32_to_f16_kernel<<<(nw + 255) / 256, 256, 0, stream>>>(Wq, Wqh, nw);
    f32_to_f16_kernel<<<(nw + 255) / 256, 256, 0, stream>>>(Wk, Wkh, nw);
    f32_to_f16_kernel<<<(nw + 255) / 256, 256, 0, stream>>>(Wv, Wvh, nw);
    f32_to_f16_kernel<<<(nw + 255) / 256, 256, 0, stream>>>(Wo, Woh, nw);
  }

  // 2) Q/K/V projections
  dim3 gblk(128), ggrd(R / 128, DM / 64);
  gemm_wmma_kernel<false><<<ggrd, gblk, 0, stream>>>(xh, Wqh, (void*)Qh, R, DM, DM);
  gemm_wmma_kernel<false><<<ggrd, gblk, 0, stream>>>(xh, Wkh, (void*)Kh, R, DM, DM);
  gemm_wmma_kernel<false><<<ggrd, gblk, 0, stream>>>(xh, Wvh, (void*)Vh, R, DM, DM);

  // 3) RoPE on Q and K
  {
    const int nr = BS * SEQ * NH * (HD / 2);  // 2^22
    rope_kernel<<<(nr + 255) / 256, 256, 0, stream>>>(Qh, tp, nr);
    rope_kernel<<<(nr + 255) / 256, 256, 0, stream>>>(Kh, tp, nr);
  }

  // 4) causal flash attention
  {
    dim3 ablk(128), agrd(SEQ / 64, NH, BS);
    flash_attn_kernel<<<agrd, ablk, 0, stream>>>(Qh, Kh, Vh, Ah);
  }

  // 5) output projection -> f32 d_out
  gemm_wmma_kernel<true><<<ggrd, gblk, 0, stream>>>(Ah, Woh, d_out, R, DM, DM);
}
